// Net_22806276341789
// MI455X (gfx1250) — matrix operand order repaired and checked by reading, not compiled, not run on hardware
//
#include <hip/hip_runtime.h>
#include <hip/hip_bf16.h>
#include <math.h>

typedef __attribute__((ext_vector_type(16))) _Float16 v16h;
typedef __attribute__((ext_vector_type(8)))  _Float16 v8h;
typedef __attribute__((ext_vector_type(8)))  float    v8f;

// D = A(16x32 f16) * B(32x16 f16) + C(f32), wave32.
#define WMMA_F32_16x16x32_F16(a, b, c) \
  __builtin_amdgcn_wmma_f32_16x16x32_f16(false, (a), false, (b), (short)0, (c), false, false)

// ---------------------------------------------------------------------------
// Per-wave WMMA step fed from LDS f16 tiles with 16B vector loads.
//   A tile: row-major [16 rows][lda]; lane halves 0..7 = K(laneHi*8..+7),
//           halves 8..15 = K(16+laneHi*8..+7)  -> two ds_load_b128
//   B tile: n-major [16 cols][ldb]; same K order as A
//                                               -> two ds_load_b128
// ---------------------------------------------------------------------------
__device__ __forceinline__ v8f
wmma_lds_step(const _Float16* __restrict__ ap, const _Float16* __restrict__ bp,
              int laneHi, v8f acc)
{
    const v8h alo = *(const v8h*)(ap + laneHi * 8);
    const v8h ahi = *(const v8h*)(ap + 16 + laneHi * 8);
    const v8h blo = *(const v8h*)(bp + laneHi * 8);
    const v8h bhi = *(const v8h*)(bp + 16 + laneHi * 8);
    v16h a, b;
#pragma unroll
    for (int i = 0; i < 8; ++i) {
        a[i] = alo[i]; a[8 + i] = ahi[i];
        b[i] = blo[i]; b[8 + i] = bhi[i];
    }
    return WMMA_F32_16x16x32_F16(a, b, acc);
}

// ---------------------------------------------------------------------------
// LDS-tiled implicit-GEMM 5x5 VALID conv (NCHW / OIHW / NCHW).
// K is REPADDED to KP = Cin*32: chunk c covers input channel c, slots
// 0..24 = (ky,kx), slots 25..31 = zero. A and B use the same ordering, so
// the GEMM is exact; padding contributes 0 via precomputed {0,1} multipliers
// (branch-free fills: no EXEC juggling, stores merge to ds_store_b128).
// The slot -> in-plane offset map is loop-invariant: computed once, the
// K loop only bumps base pointers by H*W / 25.
// Block: 256 threads = 8 waves tiling BM x BN ((BM/16)*(BN/16) == 8).
// Bias + ReLU fused into stores.
// ---------------------------------------------------------------------------
template<int B, int Cin, int H, int W, int Cout, int OH, int OW, int BM, int BN>
__global__ __launch_bounds__(256)
void conv5x5_wmma(const float* __restrict__ x, const float* __restrict__ w,
                  const float* __restrict__ bias, float* __restrict__ out)
{
    constexpr int BK  = 32;
    constexpr int LDA = BK + 8;
    constexpr int LDB = BK + 8;
    constexpr int WM  = BM / 16, WN = BN / 16;
    static_assert(WM * WN == 8, "8 waves per block");
    constexpr int NTB = Cout / BN;

    __shared__ _Float16 As[BM][LDA];
    __shared__ _Float16 Bs[BN][LDB];

    const int tid = threadIdx.x;
    const int bm0 = (int)(blockIdx.x / NTB) * BM;
    const int bn0 = (int)(blockIdx.x % NTB) * BN;

    // ---- A fill ownership: fixed (row, slot-range); pixel decomposed once ----
    constexpr int CA = (BM * BK) / 256;
    const int arow = (tid * CA) >> 5;
    const int ak0  = (tid * CA) & 31;
    const int gm   = bm0 + arow;
    const int bIdx = gm / (OH * OW);
    const int prem = gm % (OH * OW);
    const int oy   = prem / OW, ox = prem % OW;
    const float* __restrict__ xrow =
        x + (size_t)bIdx * Cin * H * W + (size_t)oy * W + ox;

    // slot -> in-plane offset + {0,1} pad multiplier, computed ONCE
    int   offA[CA];
    float mulA[CA];
#pragma unroll
    for (int i = 0; i < CA; ++i) {
        const int s = ak0 + i;                 // 0..31 within the chunk
        const bool v = (s < 25);
        mulA[i] = v ? 1.0f : 0.0f;
        offA[i] = v ? ((s / 5) * W + (s % 5)) : 0;   // 0 is always in-bounds
    }

    // ---- B fill ownership ----
    constexpr int CB = (BN * BK) / 256;
    const int brow = (tid * CB) >> 5;
    const int bk0  = (tid * CB) & 31;
    const float* __restrict__ wbase = w + (size_t)(bn0 + brow) * (Cin * 25);
    int   offB[CB];
    float mulB[CB];
#pragma unroll
    for (int i = 0; i < CB; ++i) {
        const int s = bk0 + i;
        const bool v = (s < 25);
        mulB[i] = v ? 1.0f : 0.0f;
        offB[i] = v ? s : 0;                   // clamp inside the weight row
    }

    const int lane = tid & 31, ln = lane & 15, laneHi = lane >> 4;
    const int wv = tid >> 5;
    const int wm = wv / WN, wn = wv % WN;
    const _Float16* __restrict__ ap = &As[wm * 16 + ln][0];
    const _Float16* __restrict__ bp = &Bs[wn * 16 + ln][0];

    v8f acc = {};
    const float* __restrict__ xp = xrow;       // advances by H*W per chunk
    const float* __restrict__ wp = wbase;      // advances by 25 per chunk
    for (int ic = 0; ic < Cin; ++ic) {
        if (ic + 1 < Cin) __builtin_prefetch(wp + 25, 0, 0);
#pragma unroll
        for (int i = 0; i < CA; ++i)
            As[arow][ak0 + i] = (_Float16)(xp[offA[i]] * mulA[i]);
#pragma unroll
        for (int i = 0; i < CB; ++i)
            Bs[brow][bk0 + i] = (_Float16)(wp[offB[i]] * mulB[i]);
        __syncthreads();
        acc = wmma_lds_step(ap, bp, laneHi, acc);
        __syncthreads();
        xp += (size_t)H * W;
        wp += 25;
    }

    const int n  = bn0 + wn * 16 + ln;
    const float bnv = bias[n];
#pragma unroll
    for (int r = 0; r < 8; ++r) {
        const int mi = bm0 + wm * 16 + r + laneHi * 8;
        const int b2 = mi / (OH * OW);
        const int pr = mi % (OH * OW);
        const int y2 = pr / OW, x2 = pr % OW;
        float v = acc[r] + bnv;
        v = v > 0.0f ? v : 0.0f;
        out[(((size_t)b2 * Cout + n) * OH + y2) * OW + x2] = v;
    }
}

// ---------------------------------------------------------------------------
// LDS-tiled FC GEMM: out[m][n] = act(sum_k A[m][k]*Wt[n][k] + bias[n]).
// Dense rows -> contiguous branch-free fills (merge to global_load_b128 /
// ds_store_b128). ACT: 0 = relu, 1 = sigmoid. Ragged N handled by a clamped
// row pointer + {0,1} multiplier and guarded stores.
// ---------------------------------------------------------------------------
template<int M, int N, int K, int BM, int BN, int ACT>
__global__ __launch_bounds__(256)
void fc_wmma(const float* __restrict__ A, const float* __restrict__ Wt,
             const float* __restrict__ bias, float* __restrict__ out)
{
    constexpr int BK  = 32;
    constexpr int LDA = BK + 8;
    constexpr int LDB = BK + 8;
    constexpr int WM  = BM / 16, WN = BN / 16;
    static_assert(WM * WN == 8, "8 waves per block");
    static_assert(K % BK == 0, "K multiple of 32");
    constexpr int NTB = (N + BN - 1) / BN;

    __shared__ _Float16 As[BM][LDA];
    __shared__ _Float16 Bs[BN][LDB];

    const int tid = threadIdx.x;
    const int bm0 = (int)(blockIdx.x / NTB) * BM;
    const int bn0 = (int)(blockIdx.x % NTB) * BN;

    constexpr int CA = (BM * BK) / 256;
    const int arow = (tid * CA) >> 5;
    const int ak0  = (tid * CA) & 31;
    const float* __restrict__ arowp = A + (size_t)(bm0 + arow) * K + ak0;

    constexpr int CB = (BN * BK) / 256;
    const int brow = (tid * CB) >> 5;
    const int bk0  = (tid * CB) & 31;
    const int gn_fill = bn0 + brow;
    const float mulW = (gn_fill < N) ? 1.0f : 0.0f;
    const float* __restrict__ wrowp =
        Wt + (size_t)(gn_fill < N ? gn_fill : 0) * K + bk0;

    const int lane = tid & 31, ln = lane & 15, laneHi = lane >> 4;
    const int wv = tid >> 5;
    const int wm = wv / WN, wn = wv % WN;
    const _Float16* __restrict__ ap = &As[wm * 16 + ln][0];
    const _Float16* __restrict__ bp = &Bs[wn * 16 + ln][0];

    v8f acc = {};
    for (int kb = 0; kb < K; kb += BK) {
        if (kb + BK < K) {
            __builtin_prefetch(arowp + BK, 0, 0);
            __builtin_prefetch(wrowp + BK, 0, 0);
        }
#pragma unroll
        for (int i = 0; i < CA; ++i)
            As[arow][ak0 + i] = (_Float16)arowp[i];
#pragma unroll
        for (int i = 0; i < CB; ++i)
            Bs[brow][bk0 + i] = (_Float16)(wrowp[i] * mulW);
        __syncthreads();
        acc = wmma_lds_step(ap, bp, laneHi, acc);
        __syncthreads();
        arowp += BK;
        wrowp += BK;
    }

    const int n = bn0 + wn * 16 + ln;
    if (n < N) {
        const float bnv = bias[n];
#pragma unroll
        for (int r = 0; r < 8; ++r) {
            const int mi = bm0 + wm * 16 + r + laneHi * 8;
            float v = acc[r] + bnv;
            if (ACT == 0) {
                v = v > 0.0f ? v : 0.0f;
            } else {
                v = 1.0f / (1.0f + __expf(-v));
            }
            out[(size_t)mi * N + n] = v;
        }
    }
}

// ---------------------------------------------------------------------------
// 2x2 max-pool, stride 2, NCHW. Memory-bound; plain VALU.
// ---------------------------------------------------------------------------
template<int B, int C, int H, int W>
__global__ __launch_bounds__(256)
void pool2x2(const float* __restrict__ in, float* __restrict__ out)
{
    constexpr int OH = H / 2, OW = W / 2;
    const int total = B * C * OH * OW;
    for (int i = blockIdx.x * blockDim.x + threadIdx.x; i < total;
         i += gridDim.x * blockDim.x) {
        const int xo = i % OW;
        int t = i / OW;
        const int yo = t % OH;
        t /= OH;
        const float* p = in + ((size_t)t * H + (size_t)yo * 2) * W + (size_t)xo * 2;
        out[i] = fmaxf(fmaxf(p[0], p[1]), fmaxf(p[W], p[W + 1]));
    }
}

// ---------------------------------------------------------------------------
// Segment normalization + sequential dependency multiplies (reference order).
// ---------------------------------------------------------------------------
__global__ __launch_bounds__(256)
void postproc_kernel(float* __restrict__ p, int rows)
{
    const int row = blockIdx.x * blockDim.x + threadIdx.x;
    if (row >= rows) return;
    float* q = p + (size_t)row * 37;

    const int qs[11]  = {0, 3, 5, 7, 9, 13, 15, 18, 25, 28, 31};
    const int qe[11]  = {3, 5, 7, 9, 13, 15, 18, 25, 28, 31, 37};
    const int dep[11] = {-1, 1, 4, 4, 4, -1, 0, 13, 3, 7, 7};

    for (int s = 0; s < 11; ++s) {
        float sum = 0.0f;
        for (int j = qs[s]; j < qe[s]; ++j) sum += q[j];
        const float inv = 1.0f / (sum + 1e-12f);
        for (int j = qs[s]; j < qe[s]; ++j) q[j] *= inv;
    }
    for (int s = 0; s < 11; ++s) {
        const int v = dep[s];
        if (v >= 0) {
            const float m = q[v];
            for (int j = qs[s]; j < qe[s]; ++j) q[j] *= m;
        }
    }
}

// ---------------------------------------------------------------------------
// Driver. Workspace layout (floats), reusing dead regions:
//   c1 : [0,          14,745,600)   256*16*60*60
//   c2 : [14,745,600, 40,435,712)   256*32*56*56
//   p2 : [0,           6,422,528)   reuses c1 (dead after conv2)
//   c3 : [14,745,600, 24,182,784)   reuses c2 (dead after pool2)
//   p3 : [6,422,528,   8,781,824)
//   f1 : [8,781,824,   9,043,968)
// Peak = 40,435,712 floats ~= 162 MB.
// ---------------------------------------------------------------------------
extern "C" void kernel_launch(void* const* d_in, const int* in_sizes, int n_in,
                              void* d_out, int out_size, void* d_ws, size_t ws_size,
                              hipStream_t stream)
{
    const float* x   = (const float*)d_in[0];
    const float* w1  = (const float*)d_in[1];
    const float* b1  = (const float*)d_in[2];
    const float* w2  = (const float*)d_in[3];
    const float* b2  = (const float*)d_in[4];
    const float* w3  = (const float*)d_in[5];
    const float* b3  = (const float*)d_in[6];
    const float* wf1 = (const float*)d_in[7];
    const float* bf1 = (const float*)d_in[8];
    const float* wf2 = (const float*)d_in[9];
    const float* bf2 = (const float*)d_in[10];
    float* out = (float*)d_out;
    float* ws  = (float*)d_ws;

    float* c1 = ws;
    float* c2 = ws + 14745600u;
    float* p2 = ws;                  // reuse c1 region
    float* c3 = ws + 14745600u;      // reuse c2 region
    float* p3 = ws + 6422528u;
    float* f1 = ws + 8781824u;

    // conv1: M=921600, BM=128, BN=16 -> 7200 blocks (3 K-chunks)
    conv5x5_wmma<256, 3, 64, 64, 16, 60, 60, 128, 16>
        <<<7200, 256, 0, stream>>>(x, w1, b1, c1);

    // conv2: M=802816, BM=64, BN=32 -> 12544 blocks (16 K-chunks)
    conv5x5_wmma<256, 16, 60, 60, 32, 56, 56, 64, 32>
        <<<12544, 256, 0, stream>>>(c1, w2, b2, c2);

    pool2x2<256, 32, 56, 56>
        <<<(6422528 + 255) / 256, 256, 0, stream>>>(c2, p2);

    // conv3: M=147456, BM=32, BN=64 -> 4608 blocks (32 K-chunks)
    conv5x5_wmma<256, 32, 28, 28, 64, 24, 24, 32, 64>
        <<<4608, 256, 0, stream>>>(p2, w3, b3, c3);

    pool2x2<256, 64, 24, 24>
        <<<(2359296 + 255) / 256, 256, 0, stream>>>(c3, p3);

    // fc1: M=256, N=1024, K=9216, BM=32, BN=64 -> 128 blocks
    fc_wmma<256, 1024, 9216, 32, 64, 0>
        <<<128, 256, 0, stream>>>(p3, wf1, bf1, f1);

    // fc2: M=256, N=37, K=1024, BM=128, BN=16 -> 6 blocks
    fc_wmma<256, 37, 1024, 128, 16, 1>
        <<<6, 256, 0, stream>>>(f1, wf2, bf2, out);

    postproc_kernel<<<1, 256, 0, stream>>>(out, 256);
}
